// Cross_MultiHeadedAttention_46651934769882
// MI455X (gfx1250) — matrix operand order repaired and checked by reading, not compiled, not run on hardware
//
#include <hip/hip_runtime.h>

// ---------------------------------------------------------------------------
// Cross multi-head attention with relative position bias, CDNA5 (gfx1250).
// All matmuls use V_WMMA_F32_16X16X32_BF16 (wave32, 16x16 tiles, K=32).
// ---------------------------------------------------------------------------

typedef __attribute__((ext_vector_type(16))) __bf16 v16bf;
typedef __attribute__((ext_vector_type(8)))  float  v8f;
typedef __attribute__((ext_vector_type(2)))  __bf16 bf16x2;

#define BATCH 8
#define SEQ   1024
#define FDIM  256
#define NHEAD 4
#define DK    64
#define BH    (BATCH * NHEAD)   // 32
#define MROWS (BATCH * SEQ)     // 8192

static __device__ __forceinline__ v8f wmma_bf16(v16bf a, v16bf b, v8f c) {
  // D = A(16x32) * B(32x16) + C, f32 accumulate
  return __builtin_amdgcn_wmma_f32_16x16x32_bf16(false, a, false, b,
                                                 (short)0, c, false, false);
}

// A-matrix (M=16 x K=32, bf16) per-lane fragment.  Lane l: row m = l&15,
// half = l>>4.  VGPR v holds K pair: k = 2v + (v>=4 ? 8:0) + half*8  (ISA
// 7.12.2, 16-bit A layout).  Pairs are contiguous in K -> dword loads.
static __device__ __forceinline__ v16bf load_a_bf16(const __bf16* row, int k0, int half) {
  v16bf a;
#pragma unroll
  for (int v = 0; v < 8; ++v) {
    int kk = k0 + 2 * v + ((v >= 4) ? 8 : 0) + half * 8;
    bf16x2 p = *(const bf16x2*)(row + kk);
    a[2 * v] = p.x; a[2 * v + 1] = p.y;
  }
  return a;
}

// B-matrix (K=32 x N=16, bf16).  Lane l: col n = l&15, half = l>>4.
// VGPR v holds K pair: k = 2v + (v>=4 ? 8:0) + half*8.  Source layout is B^T row-major
// (row = n, contiguous in k), so pairs are contiguous dword loads.
static __device__ __forceinline__ v16bf load_b_bf16(const __bf16* rowT, int k0, int half) {
  v16bf b;
#pragma unroll
  for (int v = 0; v < 8; ++v) {
    int kk = k0 + 2 * v + ((v >= 4) ? 8 : 0) + half * 8;
    bf16x2 p = *(const bf16x2*)(rowT + kk);
    b[2 * v] = p.x; b[2 * v + 1] = p.y;
  }
  return b;
}

// Same as load_b_bf16 but converts from an f32 source on the fly (pos_k).
static __device__ __forceinline__ v16bf load_b_f32(const float* rowT, int k0, int half) {
  v16bf b;
#pragma unroll
  for (int v = 0; v < 8; ++v) {
    int kk = k0 + 2 * v + ((v >= 4) ? 8 : 0) + half * 8;
    float2 p = *(const float2*)(rowT + kk);
    b[2 * v] = (__bf16)p.x; b[2 * v + 1] = (__bf16)p.y;
  }
  return b;
}

// ---------------------------------------------------------------------------
// Small prep kernels
// ---------------------------------------------------------------------------

// Transpose + convert a 256x256 f32 weight into bf16 W^T (wt[n][k] = w[k][n]).
__global__ __launch_bounds__(256) void wtrans_kernel(const float* __restrict__ w,
                                                     __bf16* __restrict__ wt) {
  int idx = blockIdx.x * 256 + threadIdx.x;       // 65536 elements
  int n = idx >> 8, k = idx & 255;
  wt[idx] = (__bf16)w[k * FDIM + n];
}

// Plain f32 -> bf16 convert.
__global__ __launch_bounds__(256) void cvt_kernel(const float* __restrict__ src,
                                                  __bf16* __restrict__ dst, int n) {
  int i = blockIdx.x * 256 + threadIdx.x;
  if (i < n) dst[i] = (__bf16)src[i];
}

// LayerNorm over F=256, one wave per row, output bf16.
__global__ __launch_bounds__(256) void layernorm_kernel(const float* __restrict__ x,
                                                        const float* __restrict__ g,
                                                        const float* __restrict__ bb,
                                                        __bf16* __restrict__ out) {
  int wave = threadIdx.x >> 5, lane = threadIdx.x & 31;
  int row = blockIdx.x * 8 + wave;
  const float* xr = x + (size_t)row * FDIM;
  float vals[8], s = 0.f, s2 = 0.f;
#pragma unroll
  for (int i = 0; i < 8; ++i) {
    float v = xr[lane + 32 * i];
    vals[i] = v; s += v; s2 += v * v;
  }
#pragma unroll
  for (int off = 16; off >= 1; off >>= 1) {
    s  += __shfl_xor(s,  off, 32);
    s2 += __shfl_xor(s2, off, 32);
  }
  float mu  = s * (1.0f / FDIM);
  float var = s2 * (1.0f / FDIM) - mu * mu;
  float rs  = rsqrtf(var + 1e-5f);
#pragma unroll
  for (int i = 0; i < 8; ++i) {
    int c = lane + 32 * i;
    out[(size_t)row * FDIM + c] = (__bf16)((vals[i] - mu) * rs * g[c] + bb[c]);
  }
}

// ---------------------------------------------------------------------------
// Generic projection GEMM: C[8192 x 256] = A_bf16 @ W (via W^T bf16) + bias.
// mode 0: bf16 out row-major (B,T,F)      (Q, K)
// mode 1: bf16 out transposed to (B,F,T)  (V -> V^T for the PV matmul)
// mode 2: f32 out row-major               (final @ Wo + bo)
// 8 waves/block, one 16x16 tile per wave.
// ---------------------------------------------------------------------------
__global__ __launch_bounds__(256) void gemm_kernel(const __bf16* __restrict__ A,
                                                   const __bf16* __restrict__ Wt,
                                                   const float* __restrict__ bias,
                                                   void* __restrict__ outp, int mode) {
  const int wave = threadIdx.x >> 5;
  const int lane = threadIdx.x & 31;
  const int half = lane >> 4, lm = lane & 15;
  const int tileId = blockIdx.x * 8 + wave;       // 8192 tiles
  const int mtile = tileId >> 4;                  // 512 M-tiles
  const int ntile = tileId & 15;                  // 16  N-tiles

  const __bf16* arow = A  + (size_t)(mtile * 16 + lm) * FDIM;
  const __bf16* brow = Wt + (size_t)(ntile * 16 + lm) * FDIM;

  v8f c = {};
#pragma unroll
  for (int ks = 0; ks < 8; ++ks) {
    v16bf a = load_a_bf16(arow, ks * 32, half);
    v16bf b = load_b_bf16(brow, ks * 32, half);
    c = wmma_bf16(a, b, c);
  }
  const int n = ntile * 16 + lm;
  const float bn = bias[n];

  if (mode == 0) {
    __bf16* out = (__bf16*)outp;
#pragma unroll
    for (int r = 0; r < 8; ++r) {
      int m = mtile * 16 + r + half * 8;
      out[(size_t)m * FDIM + n] = (__bf16)(c[r] + bn);
    }
  } else if (mode == 1) {
    __bf16* out = (__bf16*)outp;                  // (B, F, T)
#pragma unroll
    for (int r = 0; r < 8; ++r) {
      int m = mtile * 16 + r + half * 8;
      int b = m >> 10, t = m & 1023;
      out[((size_t)b * FDIM + n) * SEQ + t] = (__bf16)(c[r] + bn);
    }
  } else {
    float* out = (float*)outp;
#pragma unroll
    for (int r = 0; r < 8; ++r) {
      int m = mtile * 16 + r + half * 8;
      out[(size_t)m * FDIM + n] = c[r] + bn;
    }
  }
}

// ---------------------------------------------------------------------------
// Relative-position bias: for each query position q,
//   Bias[q][bh][t] = sum_d q[b,q,h*64+d] * pos_k[q,t,d]
// = GEMM (M=32 bh) x (K=64) x (N=1024 t).  pos_k read exactly once from HBM.
// grid = 1024 (q), 8 waves: mtile = wave&1, 16 N-tiles per wave.
// ---------------------------------------------------------------------------
__global__ __launch_bounds__(256) void bias_kernel(const __bf16* __restrict__ qbuf,
                                                   const float* __restrict__ pos_k,
                                                   __bf16* __restrict__ biasbuf) {
  const int q = blockIdx.x;
  const int wave = threadIdx.x >> 5;
  const int lane = threadIdx.x & 31;
  const int half = lane >> 4, lm = lane & 15;
  const int mtile  = wave & 1;
  const int ngroup = wave >> 1;

  const int bh = mtile * 16 + lm;                 // A row = (b,h) pair
  const int bb = bh >> 2, hh = bh & 3;
  const __bf16* arow = qbuf + ((size_t)(bb * SEQ + q)) * FDIM + hh * DK;
  v16bf a0 = load_a_bf16(arow, 0,  half);
  v16bf a1 = load_a_bf16(arow, 32, half);

  const float* pq = pos_k + (size_t)q * SEQ * DK;
#pragma unroll 1
  for (int j = 0; j < 16; ++j) {
    int nt = ngroup * 16 + j;
    int t  = nt * 16 + lm;
    const float* brow = pq + (size_t)t * DK;
    __builtin_prefetch(brow + 16 * DK, 0, 1);     // global_prefetch next tile
    v16bf b0 = load_b_f32(brow, 0,  half);
    v16bf b1 = load_b_f32(brow, 32, half);
    v8f c = {};
    c = wmma_bf16(a0, b0, c);
    c = wmma_bf16(a1, b1, c);
#pragma unroll
    for (int r = 0; r < 8; ++r) {
      int obh = mtile * 16 + r + half * 8;
      biasbuf[((size_t)q * BH + obh) * SEQ + t] = (__bf16)c[r];
    }
  }
}

// ---------------------------------------------------------------------------
// Fused attention: one block = (16 q-rows, batch b), one wave per head.
// Flash-style online softmax over T2=1024 in 64-wide key blocks.
//   scores = (Q K^T + Bias) / 8, masked -> -inf; P @ V accumulated in f32.
// ---------------------------------------------------------------------------
__global__ __launch_bounds__(128) void attn_kernel(const __bf16* __restrict__ qbuf,
                                                   const __bf16* __restrict__ kbuf,
                                                   const __bf16* __restrict__ vt,
                                                   const __bf16* __restrict__ biasbuf,
                                                   const int* __restrict__ mask,
                                                   __bf16* __restrict__ attnbuf) {
  __shared__ __bf16 Pbuf[NHEAD * 16 * 64];

  const int h    = threadIdx.x >> 5;              // head = wave
  const int lane = threadIdx.x & 31;
  const int half = lane >> 4, lm = lane & 15;
  const int qt = blockIdx.x;                      // 0..63
  const int b  = blockIdx.y;                      // 0..7
  const int q0 = qt * 16;
  const int bh = b * NHEAD + h;

  const __bf16* aq = qbuf + ((size_t)(b * SEQ + q0 + lm)) * FDIM + h * DK;
  v16bf qa0 = load_a_bf16(aq, 0,  half);
  v16bf qa1 = load_a_bf16(aq, 32, half);

  v8f O[4] = {v8f{}, v8f{}, v8f{}, v8f{}};
  float mrow[8], lrow[8];
#pragma unroll
  for (int r = 0; r < 8; ++r) { mrow[r] = -1e30f; lrow[r] = 0.f; }

  __bf16* P = Pbuf + h * 16 * 64;

#pragma unroll 1
  for (int tb = 0; tb < SEQ; tb += 64) {
    float s[4][8];
#pragma unroll
    for (int j = 0; j < 4; ++j) {
      int t0 = tb + j * 16;
      const __bf16* krow = kbuf + ((size_t)(b * SEQ + t0 + lm)) * FDIM + h * DK;
      v16bf kb0 = load_b_bf16(krow, 0,  half);
      v16bf kb1 = load_b_bf16(krow, 32, half);
      v8f acc = {};
      acc = wmma_bf16(qa0, kb0, acc);
      acc = wmma_bf16(qa1, kb1, acc);
#pragma unroll
      for (int r = 0; r < 8; ++r) {
        int qg = q0 + r + half * 8;
        float bias = (float)biasbuf[((size_t)qg * BH + bh) * SEQ + t0 + lm];
        int mk = mask[((size_t)b * SEQ + qg) * SEQ + t0 + lm];
        s[j][r] = (mk == 0) ? -1e30f : (acc[r] + bias) * 0.125f;
      }
    }
    // --- online softmax update (row = r + half*8, 16 cols across half-wave)
#pragma unroll
    for (int r = 0; r < 8; ++r) {
      float mx = fmaxf(fmaxf(s[0][r], s[1][r]), fmaxf(s[2][r], s[3][r]));
#pragma unroll
      for (int off = 8; off >= 1; off >>= 1) mx = fmaxf(mx, __shfl_xor(mx, off, 16));
      float mnew = fmaxf(mrow[r], mx);
      float corr = __expf(mrow[r] - mnew);
      float rsum = 0.f;
#pragma unroll
      for (int j = 0; j < 4; ++j) {
        float p = (s[j][r] <= -1e29f) ? 0.f : __expf(s[j][r] - mnew);
        s[j][r] = p;
        rsum += p;
      }
#pragma unroll
      for (int off = 8; off >= 1; off >>= 1) rsum += __shfl_xor(rsum, off, 16);
      lrow[r] = lrow[r] * corr + rsum;
      mrow[r] = mnew;
#pragma unroll
      for (int nt = 0; nt < 4; ++nt) O[nt][r] *= corr;
#pragma unroll
      for (int j = 0; j < 4; ++j)
        P[(r + half * 8) * 64 + j * 16 + lm] = (__bf16)s[j][r];
    }
    // P (16x64) as A fragments from LDS (same-wave DS ops are in-order)
    const __bf16* prow = P + lm * 64;
    v16bf pa0 = load_a_bf16(prow, 0,  half);
    v16bf pa1 = load_a_bf16(prow, 32, half);
    // P @ V : V stored transposed (B,F,T) so B-pairs are contiguous in t
#pragma unroll
    for (int nt = 0; nt < 4; ++nt) {
      const __bf16* vrow = vt + ((size_t)(b * FDIM + h * DK + nt * 16 + lm)) * SEQ + tb;
      v16bf vb0 = load_b_bf16(vrow, 0,  half);
      v16bf vb1 = load_b_bf16(vrow, 32, half);
      O[nt] = wmma_bf16(pa0, vb0, O[nt]);
      O[nt] = wmma_bf16(pa1, vb1, O[nt]);
    }
  }
  // epilogue: normalize and store (B,T,F) bf16 for the Wo projection
#pragma unroll
  for (int r = 0; r < 8; ++r) {
    float inv = (lrow[r] > 0.f) ? 1.0f / lrow[r] : 0.f;
    int qg = q0 + r + half * 8;
    __bf16* orow = attnbuf + ((size_t)(b * SEQ + qg)) * FDIM + h * DK;
#pragma unroll
    for (int nt = 0; nt < 4; ++nt)
      orow[nt * 16 + lm] = (__bf16)(O[nt][r] * inv);
  }
}

// ---------------------------------------------------------------------------
// Host launch
// ---------------------------------------------------------------------------
extern "C" void kernel_launch(void* const* d_in, const int* in_sizes, int n_in,
                              void* d_out, int out_size, void* d_ws, size_t ws_size,
                              hipStream_t stream) {
  (void)in_sizes; (void)n_in; (void)out_size; (void)ws_size;
  const float* x     = (const float*)d_in[0];
  const float* q_in  = (const float*)d_in[1];
  const float* pos_k = (const float*)d_in[2];
  const int*   mask  = (const int*)d_in[3];
  const float* ln_g  = (const float*)d_in[4];
  const float* ln_b  = (const float*)d_in[5];
  const float* Wq    = (const float*)d_in[6];
  const float* bq    = (const float*)d_in[7];
  const float* Wk    = (const float*)d_in[8];
  const float* bk    = (const float*)d_in[9];
  const float* Wv    = (const float*)d_in[10];
  const float* bv    = (const float*)d_in[11];
  const float* Wo    = (const float*)d_in[12];
  const float* bo    = (const float*)d_in[13];

  // Workspace carve-up (~89 MB total)
  char* ws = (char*)d_ws;
  size_t off = 0;
  auto carve = [&](size_t bytes) {
    char* p = ws + off;
    off += (bytes + 255) & ~(size_t)255;
    return p;
  };
  __bf16* wtq     = (__bf16*)carve((size_t)FDIM * FDIM * 2);
  __bf16* wtk     = (__bf16*)carve((size_t)FDIM * FDIM * 2);
  __bf16* wtv     = (__bf16*)carve((size_t)FDIM * FDIM * 2);
  __bf16* wto     = (__bf16*)carve((size_t)FDIM * FDIM * 2);
  __bf16* qin_bf  = (__bf16*)carve((size_t)MROWS * FDIM * 2);
  __bf16* xn_bf   = (__bf16*)carve((size_t)MROWS * FDIM * 2);
  __bf16* qbuf    = (__bf16*)carve((size_t)MROWS * FDIM * 2);
  __bf16* kbuf    = (__bf16*)carve((size_t)MROWS * FDIM * 2);
  __bf16* vtbuf   = (__bf16*)carve((size_t)MROWS * FDIM * 2);   // (B,F,T)
  __bf16* attnbuf = (__bf16*)carve((size_t)MROWS * FDIM * 2);
  __bf16* biasbuf = (__bf16*)carve((size_t)SEQ * BH * SEQ * 2); // 64 MB

  // 1) weights -> bf16 transposed
  wtrans_kernel<<<256, 256, 0, stream>>>(Wq, wtq);
  wtrans_kernel<<<256, 256, 0, stream>>>(Wk, wtk);
  wtrans_kernel<<<256, 256, 0, stream>>>(Wv, wtv);
  wtrans_kernel<<<256, 256, 0, stream>>>(Wo, wto);
  // 2) q_in -> bf16 ; layernorm(x) -> bf16
  cvt_kernel<<<MROWS * FDIM / 256, 256, 0, stream>>>(q_in, qin_bf, MROWS * FDIM);
  layernorm_kernel<<<MROWS / 8, 256, 0, stream>>>(x, ln_g, ln_b, xn_bf);
  // 3) projections (WMMA)
  gemm_kernel<<<1024, 256, 0, stream>>>(qin_bf, wtq, bq, (void*)qbuf, 0);
  gemm_kernel<<<1024, 256, 0, stream>>>(xn_bf,  wtk, bk, (void*)kbuf, 0);
  gemm_kernel<<<1024, 256, 0, stream>>>(xn_bf,  wtv, bv, (void*)vtbuf, 1);
  // 4) relative-position bias (WMMA, pos_k read once)
  bias_kernel<<<SEQ, 256, 0, stream>>>(qbuf, pos_k, biasbuf);
  // 5) fused flash attention (WMMA)
  attn_kernel<<<dim3(SEQ / 16, BATCH), 128, 0, stream>>>(qbuf, kbuf, vtbuf,
                                                         biasbuf, mask, attnbuf);
  // 6) output projection -> f32 d_out
  gemm_kernel<<<1024, 256, 0, stream>>>(attnbuf, wto, bo, d_out, 2);
}
